// Decoder_1142461301311
// MI455X (gfx1250) — matrix operand order repaired and checked by reading, not compiled, not run on hardware
//
#include <hip/hip_runtime.h>
#include <hip/hip_bf16.h>

// ---------------------------------------------------------------------------
// Shapes (hardcoded from the reference): B=64, S=400, H=512, E=256, V=50000,
// OOV=100. 2H=1024, 4H=2048, 2H+E=1280, 3H=1536, 4H+E=2304, V+OOV=50100.
// ---------------------------------------------------------------------------
#define BB    64
#define SS    400
#define HH    512
#define EE    256
#define VV    50000
#define OOVV  100
#define TWOH  1024
#define LDVD  50100

typedef __attribute__((ext_vector_type(16))) __bf16 v16bf;
typedef __attribute__((ext_vector_type(8)))  __bf16 v8bf;
typedef __attribute__((ext_vector_type(8)))  float  v8f;

__device__ __forceinline__ float sigmoidf_(float x) { return 1.f / (1.f + __expf(-x)); }
__device__ __forceinline__ float tanh_hw(float x)   { return __builtin_amdgcn_tanhf(x); }

// ---------------------------------------------------------------------------
// K0: pack bf16 inputs: ctxin = bf16([context(1024) | emb[tok](256)]) [64,1280]
// and h0_bf [64,512]. grid 64, block 256
// ---------------------------------------------------------------------------
__global__ void k_pack_in(const int* __restrict__ sin, const float* __restrict__ emb,
                          const float* __restrict__ context, const float* __restrict__ h0,
                          __bf16* __restrict__ ctxin_bf, __bf16* __restrict__ h0_bf) {
  int b = blockIdx.x, tid = threadIdx.x;
  int tok = sin[b];
  for (int k = tid; k < TWOH; k += 256)
    ctxin_bf[(size_t)b * (TWOH + EE) + k] = (__bf16)context[(size_t)b * TWOH + k];
  ctxin_bf[(size_t)b * (TWOH + EE) + TWOH + tid] = (__bf16)emb[(size_t)tok * EE + tid];
  for (int k = tid; k < HH; k += 256)
    h0_bf[(size_t)b * HH + k] = (__bf16)h0[(size_t)b * HH + k];
}

// ---------------------------------------------------------------------------
// Generic WMMA GEMM: D[64,N] (+)= A_bf16[64,K] @ W_f32[N,K]^T + bias[N]
// A is bf16 row-major; W rows converted to bf16 on load (streamed once).
// One wave per 16-wide N tile; computes all four 16-row M tiles so the
// B operand is reused 4x. K multiple of 32, N multiple of 16.
// accum!=0: preload C from D (used to split the LSTM gate GEMM in two).
// A/B lanes hold contiguous k = 16*(lane/16) + s (same in both); f32 C/D layout per CDNA5 ISA 7.12.2.
// ---------------------------------------------------------------------------
__global__ __launch_bounds__(256)
void k_gemm_bf16(const __bf16* __restrict__ A, const float* __restrict__ W,
                 const float* __restrict__ bias, float* __restrict__ D,
                 __bf16* __restrict__ Dbf, int N, int K, int ldD, int accum) {
  int wave = blockIdx.x * (blockDim.x >> 5) + (threadIdx.x >> 5);
  int lane = threadIdx.x & 31;
  int ntiles = N >> 4;
  if (wave >= ntiles) return;
  int n0  = wave << 4;
  int col = lane & 15;   // N column within tile (and M row within tile for A)
  int hi  = lane >> 4;   // lane-half selector per ISA layout tables

  const float*  wrow  = W + (size_t)(n0 + col) * K + hi * 16; // 16 contiguous K
  const __bf16* abase = A + (size_t)col * K + hi * 16;        // 16 contiguous K

  v8f acc[4] = {};
  if (accum) {
#pragma unroll
    for (int t = 0; t < 4; ++t)
#pragma unroll
      for (int r = 0; r < 8; ++r)
        acc[t][r] = D[(size_t)(t * 16 + hi * 8 + r) * ldD + n0 + col];
  }
  for (int k0 = 0; k0 < K; k0 += 32) {
    __builtin_prefetch(wrow + k0 + 32, 0, 1);  // global_prefetch_b8, next K step
    // ---- B tile (32x16): lane holds 16 contiguous f32 -> bf16 ----
    const float4* w4 = (const float4*)(wrow + k0);
    float wf[16];
    *(float4*)&wf[0]  = w4[0];
    *(float4*)&wf[4]  = w4[1];
    *(float4*)&wf[8]  = w4[2];
    *(float4*)&wf[12] = w4[3];
    v16bf bm;
#pragma unroll
    for (int i = 0; i < 16; ++i) bm[i] = (__bf16)wf[i];
    // ---- 4 A tiles (16x32 each) + WMMA ----
#pragma unroll
    for (int t = 0; t < 4; ++t) {
      const __bf16* ap = abase + (size_t)t * 16 * K + k0;
      v8bf lo  = *(const v8bf*)(ap);       // K = k0 + hi*16     .. +7
      v8bf hi8 = *(const v8bf*)(ap + 8);   // K = k0 + hi*16 + 8 .. +15
      v16bf am = __builtin_shufflevector(lo, hi8, 0, 1, 2, 3, 4, 5, 6, 7,
                                         8, 9, 10, 11, 12, 13, 14, 15);
      acc[t] = __builtin_amdgcn_wmma_f32_16x16x32_bf16(
          false, am, false, bm, (short)0, acc[t], false, false);
    }
  }
  // ---- store: C/D layout — VGPR r, lane<16: M=r ; lane>=16: M=r+8 ----
  float bb = bias[n0 + col];
#pragma unroll
  for (int t = 0; t < 4; ++t)
#pragma unroll
    for (int r = 0; r < 8; ++r)
      D[(size_t)(t * 16 + hi * 8 + r) * ldD + n0 + col] = acc[t][r] + bb;
  if (Dbf) {
#pragma unroll
    for (int t = 0; t < 4; ++t)
#pragma unroll
      for (int r = 0; r < 8; ++r)
        Dbf[(size_t)(t * 16 + hi * 8 + r) * N + n0 + col] = (__bf16)(acc[t][r] + bb);
  }
}

// ---------------------------------------------------------------------------
// WMMA GEMM with LDS-staged A (for logits: N=50000, K=512).
// Every wave needs ALL of A (64x512 bf16 = 64 KB); stage it once per block
// into LDS with CDNA5 async load-to-LDS (ASYNCcnt), then feed WMMA A operands
// from ds_load_b128 while W streams from global (HBM, read once).
// ---------------------------------------------------------------------------
#define KLOG 512
__global__ __launch_bounds__(256)
void k_gemm_bf16_lds(const __bf16* __restrict__ A, const float* __restrict__ W,
                     const float* __restrict__ bias, float* __restrict__ D,
                     int N, int ldD) {
  __shared__ __bf16 As[BB * KLOG];                  // 64 KB
  // ---- cooperative async stage: 256 threads x 16 chunks x 16B = 64 KB ----
  {
    unsigned lds_base = (unsigned)(size_t)(&As[0]); // addr[31:0] = LDS offset
    const char* Ab = (const char*)A;
#pragma unroll
    for (int j = 0; j < 16; ++j) {
      unsigned off = (threadIdx.x + j * 256) * 16;  // byte offset
      unsigned lds_off = lds_base + off;
      asm volatile("global_load_async_to_lds_b128 %0, %1, %2"
                   :: "v"(lds_off), "v"(off), "s"(Ab) : "memory");
    }
    asm volatile("s_wait_asynccnt 0" ::: "memory");
  }
  __syncthreads();

  int wave = blockIdx.x * (blockDim.x >> 5) + (threadIdx.x >> 5);
  int lane = threadIdx.x & 31;
  int ntiles = N >> 4;
  if (wave < ntiles) {
    int n0  = wave << 4;
    int col = lane & 15;
    int hi  = lane >> 4;
    const float*  wrow  = W + (size_t)(n0 + col) * KLOG + hi * 16;
    const __bf16* abase = As + col * KLOG + hi * 16; // LDS
    v8f acc[4] = {};
    for (int k0 = 0; k0 < KLOG; k0 += 32) {
      __builtin_prefetch(wrow + k0 + 32, 0, 1);
      const float4* w4 = (const float4*)(wrow + k0);
      float wf[16];
      *(float4*)&wf[0]  = w4[0];
      *(float4*)&wf[4]  = w4[1];
      *(float4*)&wf[8]  = w4[2];
      *(float4*)&wf[12] = w4[3];
      v16bf bm;
#pragma unroll
      for (int i = 0; i < 16; ++i) bm[i] = (__bf16)wf[i];
#pragma unroll
      for (int t = 0; t < 4; ++t) {
        const __bf16* ap = abase + t * 16 * KLOG + k0;
        v8bf lo  = *(const v8bf*)(ap);       // ds_load_b128
        v8bf hi8 = *(const v8bf*)(ap + 8);   // ds_load_b128
        v16bf am = __builtin_shufflevector(lo, hi8, 0, 1, 2, 3, 4, 5, 6, 7,
                                           8, 9, 10, 11, 12, 13, 14, 15);
        acc[t] = __builtin_amdgcn_wmma_f32_16x16x32_bf16(
            false, am, false, bm, (short)0, acc[t], false, false);
      }
    }
    float bb = bias[n0 + col];
#pragma unroll
    for (int t = 0; t < 4; ++t)
#pragma unroll
      for (int r = 0; r < 8; ++r)
        D[(size_t)(t * 16 + hi * 8 + r) * ldD + n0 + col] = acc[t][r] + bb;
  }
}

// ---------------------------------------------------------------------------
// K2: elementwise LSTM cell from precomputed gates (order i,f,g,o).
// Writes hidden_out=[h|c] (f32 in d_out) and a bf16 copy. grid 64, block 512
// ---------------------------------------------------------------------------
__global__ __launch_bounds__(512)
void k_lstm_elem(const float* __restrict__ gates, const float* __restrict__ c0,
                 float* __restrict__ hid, __bf16* __restrict__ hid_bf) {
  int b = blockIdx.x, n = threadIdx.x;
  const float* g = gates + (size_t)b * 4 * HH;
  float gi = g[n], gf = g[n + HH], gg = g[n + 2 * HH], go = g[n + 3 * HH];
  float c = sigmoidf_(gf) * c0[(size_t)b * HH + n] + sigmoidf_(gi) * tanh_hw(gg);
  float h = sigmoidf_(go) * tanh_hw(c);
  hid[(size_t)b * TWOH + n] = h;
  hid[(size_t)b * TWOH + HH + n] = c;
  hid_bf[(size_t)b * TWOH + n] = (__bf16)h;
  hid_bf[(size_t)b * TWOH + HH + n] = (__bf16)c;
}

// ---------------------------------------------------------------------------
// K4: attention energies. One wave per (s,b) row of 1024, float4 loads,
// hardware v_tanh_f32. grid 3200, block 256 (8 waves).
// Streams encoder_features (104.8 MB).
// ---------------------------------------------------------------------------
__global__ void k_energy(const float* __restrict__ ef, const float* __restrict__ dec_feat,
                         const float* __restrict__ coverage, const float* __restrict__ W_cov,
                         const float* __restrict__ b_cov, const float* __restrict__ w_att,
                         const float* __restrict__ b_att, float* __restrict__ e_ws) {
  int row  = blockIdx.x * 8 + (threadIdx.x >> 5);   // s*64 + b, < 25600
  int lane = threadIdx.x & 31;
  int b = row & (BB - 1);
  float cov = coverage[row];
  const float4* er = (const float4*)(ef + (size_t)row * TWOH);
  const float4* dr = (const float4*)(dec_feat + (size_t)b * TWOH);
  const float4* wc = (const float4*)W_cov;
  const float4* bc = (const float4*)b_cov;
  const float4* wa = (const float4*)w_att;
  float p = 0.f;
#pragma unroll
  for (int j = 0; j < 8; ++j) {
    int i = j * 32 + lane;                          // 256 float4 per row
    float4 e4 = er[i], d4 = dr[i], w4 = wc[i], b4 = bc[i], a4 = wa[i];
    p += tanh_hw(e4.x + d4.x + cov * w4.x + b4.x) * a4.x;
    p += tanh_hw(e4.y + d4.y + cov * w4.y + b4.y) * a4.y;
    p += tanh_hw(e4.z + d4.z + cov * w4.z + b4.z) * a4.z;
    p += tanh_hw(e4.w + d4.w + cov * w4.w + b4.w) * a4.w;
  }
  for (int off = 16; off > 0; off >>= 1) p += __shfl_down(p, off, 32);
  if (lane == 0) e_ws[row] = p + b_att[0];
}

// ---------------------------------------------------------------------------
// K5: masked softmax over S per batch column + coverage update.
// a = exp(e-max)*mask / sum(exp(e-max)*mask); cov_next = cov + a
// grid 64, block 256
// ---------------------------------------------------------------------------
__global__ void k_softmax_a(const float* __restrict__ e, const float* __restrict__ mask,
                            const float* __restrict__ coverage, float* __restrict__ a_out,
                            float* __restrict__ covn) {
  int b = blockIdx.x, tid = threadIdx.x;
  __shared__ float sd[256];
  float m = -3.4e38f;
  for (int s = tid; s < SS; s += 256) m = fmaxf(m, e[s * BB + b]);
  sd[tid] = m; __syncthreads();
  for (int st = 128; st > 0; st >>= 1) { if (tid < st) sd[tid] = fmaxf(sd[tid], sd[tid + st]); __syncthreads(); }
  m = sd[0]; __syncthreads();
  float sum = 0.f;
  for (int s = tid; s < SS; s += 256) {
    float ex = __expf(e[s * BB + b] - m) * mask[s * BB + b];
    a_out[s * BB + b] = ex;
    sum += ex;
  }
  sd[tid] = sum; __syncthreads();
  for (int st = 128; st > 0; st >>= 1) { if (tid < st) sd[tid] += sd[tid + st]; __syncthreads(); }
  float inv = 1.f / sd[0];
  for (int s = tid; s < SS; s += 256) {
    float aa = a_out[s * BB + b] * inv;
    a_out[s * BB + b] = aa;
    covn[s * BB + b] = coverage[s * BB + b] + aa;
  }
}

// ---------------------------------------------------------------------------
// K6a/K6b: ctx_new[b,d] = sum_s a[s,b]*encoder_out[b,s,d] (deterministic,
// two-stage, float4). Streams encoder_out (104.8 MB).
// ---------------------------------------------------------------------------
__global__ void k_ctx_partial(const float* __restrict__ a_out, const float* __restrict__ eo,
                              float* __restrict__ part) {
  int b = blockIdx.x, ch = blockIdx.y, tid = threadIdx.x;  // 64 x 8, 256
  float4 acc = {0.f, 0.f, 0.f, 0.f};
  int s0 = ch * (SS / 8);
  for (int s = s0; s < s0 + SS / 8; ++s) {
    float av = a_out[s * BB + b];
    float4 e4 = ((const float4*)(eo + ((size_t)b * SS + s) * TWOH))[tid];
    acc.x += av * e4.x; acc.y += av * e4.y; acc.z += av * e4.z; acc.w += av * e4.w;
  }
  ((float4*)(part + ((size_t)ch * BB + b) * TWOH))[tid] = acc;
}

__global__ void k_ctx_reduce(const float* __restrict__ part, float* __restrict__ ctxn) {
  int idx = blockIdx.x * 256 + threadIdx.x;  // 16384 float4
  float4 s = {0.f, 0.f, 0.f, 0.f};
  for (int ch = 0; ch < 8; ++ch) {
    float4 p = ((const float4*)part)[(size_t)ch * (BB * TWOH / 4) + idx];
    s.x += p.x; s.y += p.y; s.z += p.z; s.w += p.w;
  }
  ((float4*)ctxn)[idx] = s;
}

// ---------------------------------------------------------------------------
// K7: p_gen[b] = sigmoid([ctx_new|hidden|ctx_out].W_pg + b_pg); also pack
// A2_bf16 = bf16([h | ctx_new]) for the W_o1 GEMM. grid 64, block 256
// ---------------------------------------------------------------------------
__global__ void k_pgen_pack(const float* __restrict__ ctxn, const float* __restrict__ hid,
                            const float* __restrict__ ctx_out, const float* __restrict__ W_pg,
                            const float* __restrict__ b_pg, float* __restrict__ pgen,
                            __bf16* __restrict__ a2) {
  int b = blockIdx.x, tid = threadIdx.x;
  __shared__ float sd[256];
  float s = 0.f;
  for (int k = tid; k < TWOH; k += 256) s += ctxn[(size_t)b * TWOH + k] * W_pg[k];
  for (int k = tid; k < TWOH; k += 256) s += hid[(size_t)b * TWOH + k] * W_pg[TWOH + k];
  s += ctx_out[(size_t)b * EE + tid] * W_pg[2 * TWOH + tid];
  sd[tid] = s; __syncthreads();
  for (int st = 128; st > 0; st >>= 1) { if (tid < st) sd[tid] += sd[tid + st]; __syncthreads(); }
  if (tid == 0) pgen[b] = sigmoidf_(sd[0] + b_pg[0]);
  for (int k = tid; k < HH; k += 256)
    a2[(size_t)b * (3 * HH) + k] = (__bf16)hid[(size_t)b * TWOH + k];
  for (int k = tid; k < TWOH; k += 256)
    a2[(size_t)b * (3 * HH) + HH + k] = (__bf16)ctxn[(size_t)b * TWOH + k];
}

// ---------------------------------------------------------------------------
// K10: vocab softmax row b over 50000 logits (in-place in vd, float4),
// scale by p_gen, zero the OOV tail. grid 64, block 256.
// Row base b*50100 floats is 16B-aligned; 50000 = 12500 float4 exactly.
// ---------------------------------------------------------------------------
__global__ void k_softmax_v(float* __restrict__ vd, const float* __restrict__ pgen) {
  int b = blockIdx.x, tid = threadIdx.x;
  float4* row = (float4*)(vd + (size_t)b * LDVD);
  __shared__ float sd[256];
  float m = -3.4e38f;
  for (int i = tid; i < VV / 4; i += 256) {
    float4 v = row[i];
    m = fmaxf(m, fmaxf(fmaxf(v.x, v.y), fmaxf(v.z, v.w)));
  }
  sd[tid] = m; __syncthreads();
  for (int st = 128; st > 0; st >>= 1) { if (tid < st) sd[tid] = fmaxf(sd[tid], sd[tid + st]); __syncthreads(); }
  m = sd[0]; __syncthreads();
  float sum = 0.f;
  for (int i = tid; i < VV / 4; i += 256) {
    float4 v = row[i];
    v.x = __expf(v.x - m); v.y = __expf(v.y - m); v.z = __expf(v.z - m); v.w = __expf(v.w - m);
    sum += (v.x + v.y) + (v.z + v.w);
    row[i] = v;
  }
  sd[tid] = sum; __syncthreads();
  for (int st = 128; st > 0; st >>= 1) { if (tid < st) sd[tid] += sd[tid + st]; __syncthreads(); }
  float scale = pgen[b] / sd[0];
  for (int i = tid; i < VV / 4; i += 256) {
    float4 v = row[i];
    v.x *= scale; v.y *= scale; v.z *= scale; v.w *= scale;
    row[i] = v;
  }
  float* rowf = (float*)row;
  for (int i = VV + tid; i < LDVD; i += 256) rowf[i] = 0.f;
}

// ---------------------------------------------------------------------------
// K11: copy-distribution scatter-add: vd[b, texts[s,b]] += (1-p_gen[b])*a[s,b]
// ---------------------------------------------------------------------------
__global__ void k_scatter(const float* __restrict__ a_out, const int* __restrict__ texts,
                          const float* __restrict__ pgen, float* __restrict__ vd) {
  int idx = blockIdx.x * 256 + threadIdx.x;
  if (idx >= SS * BB) return;
  int b = idx & (BB - 1);
  float w = (1.f - pgen[b]) * a_out[idx];
  atomicAdd(&vd[(size_t)b * LDVD + texts[idx]], w);
}

// ---------------------------------------------------------------------------
extern "C" void kernel_launch(void* const* d_in, const int* in_sizes, int n_in,
                              void* d_out, int out_size, void* d_ws, size_t ws_size,
                              hipStream_t stream) {
  const int*   summaries_in = (const int*)d_in[0];
  const float* h0      = (const float*)d_in[1];
  const float* c0      = (const float*)d_in[2];
  const float* enc_out = (const float*)d_in[3];
  const float* enc_f   = (const float*)d_in[4];
  const float* enc_m   = (const float*)d_in[5];
  const float* context = (const float*)d_in[6];
  const float* cover   = (const float*)d_in[7];
  const int*   texts   = (const int*)d_in[8];
  const float* emb     = (const float*)d_in[10];
  const float* W_feat  = (const float*)d_in[11];
  const float* b_feat  = (const float*)d_in[12];
  const float* W_cov   = (const float*)d_in[13];
  const float* b_cov   = (const float*)d_in[14];
  const float* w_att   = (const float*)d_in[15];
  const float* b_att   = (const float*)d_in[16];
  const float* W_ih    = (const float*)d_in[17];
  const float* W_hh    = (const float*)d_in[18];
  const float* b_ih    = (const float*)d_in[19];
  const float* b_hh    = (const float*)d_in[20];
  const float* W_ctx   = (const float*)d_in[21];
  const float* b_ctx   = (const float*)d_in[22];
  const float* W_pg    = (const float*)d_in[23];
  const float* b_pg    = (const float*)d_in[24];
  const float* W_o1    = (const float*)d_in[25];
  const float* b_o1    = (const float*)d_in[26];
  const float* W_o2    = (const float*)d_in[27];
  const float* b_o2    = (const float*)d_in[28];

  // d_out layout: vd[64,50100] | hidden[64,1024] | ctx_new[64,1024] | a[400,64] | cov[400,64]
  float* out   = (float*)d_out;
  float* vd    = out;
  float* hid   = out + (size_t)BB * LDVD;
  float* ctxn  = hid + BB * TWOH;
  float* a_out = ctxn + BB * TWOH;
  float* covn  = a_out + SS * BB;

  // workspace layout
  char* w = (char*)d_ws;
  __bf16* ctxin_bf  = (__bf16*)w; w += (size_t)BB * (TWOH + EE) * 2; // 64x1280 bf16
  __bf16* h0_bf     = (__bf16*)w; w += (size_t)BB * HH * 2;          // 64x512  bf16
  float*  ctx_out_f = (float*)w;  w += (size_t)BB * EE * 4;          // 64x256  f32
  __bf16* ctx_out_bf= (__bf16*)w; w += (size_t)BB * EE * 2;          // 64x256  bf16
  float*  gates     = (float*)w;  w += (size_t)BB * 4 * HH * 4;      // 64x2048 f32
  __bf16* hid_bf    = (__bf16*)w; w += (size_t)BB * TWOH * 2;        // 64x1024 bf16
  float*  dec_feat  = (float*)w;  w += (size_t)BB * TWOH * 4;        // 64x1024 f32
  float*  e_ws      = (float*)w;  w += (size_t)SS * BB * 4;          // 400x64  f32
  __bf16* a2_bf     = (__bf16*)w; w += (size_t)BB * 3 * HH * 2;      // 64x1536 bf16
  float*  out1_f    = (float*)w;  w += (size_t)BB * HH * 4;          // 64x512  f32
  __bf16* out1_bf   = (__bf16*)w; w += (size_t)BB * HH * 2;          // 64x512  bf16
  float*  pgen      = (float*)w;  w += 256;                          // 64 f32
  float*  part_ctx  = (float*)w;  w += (size_t)8 * BB * TWOH * 4;    // 8x64x1024 f32

  // 0. pack bf16 inputs (embedding gather + context + h0)
  k_pack_in<<<BB, 256, 0, stream>>>(summaries_in, emb, context, h0, ctxin_bf, h0_bf);
  // 1. ctx_out = [context|x] @ W_ctx^T + b_ctx   (WMMA, N=256, K=1280)
  k_gemm_bf16<<<2, 256, 0, stream>>>(ctxin_bf, W_ctx, b_ctx, ctx_out_f, ctx_out_bf,
                                     EE, TWOH + EE, EE, 0);
  // 2a. gates = ctx_out @ W_ih^T + b_ih          (WMMA, N=2048, K=256)
  k_gemm_bf16<<<16, 256, 0, stream>>>(ctx_out_bf, W_ih, b_ih, gates, (__bf16*)nullptr,
                                      4 * HH, EE, 4 * HH, 0);
  // 2b. gates += h0 @ W_hh^T + b_hh              (WMMA accumulate, N=2048, K=512)
  k_gemm_bf16<<<16, 256, 0, stream>>>(h0_bf, W_hh, b_hh, gates, (__bf16*)nullptr,
                                      4 * HH, HH, 4 * HH, 1);
  // 2c. elementwise LSTM -> hidden_out (f32 + bf16 copy)
  k_lstm_elem<<<BB, HH, 0, stream>>>(gates, c0, hid, hid_bf);
  // 3. dec_feat = hidden @ W_feat^T + b_feat     (WMMA, N=1024, K=1024)
  k_gemm_bf16<<<8, 256, 0, stream>>>(hid_bf, W_feat, b_feat, dec_feat, (__bf16*)nullptr,
                                     TWOH, TWOH, TWOH, 0);
  // 4. attention energies (streams encoder_features, 104.8 MB, b128 loads)
  k_energy<<<SS * BB / 8, 256, 0, stream>>>(enc_f, dec_feat, cover, W_cov, b_cov,
                                            w_att, b_att, e_ws);
  // 5. masked softmax over S + coverage update
  k_softmax_a<<<BB, 256, 0, stream>>>(e_ws, enc_m, cover, a_out, covn);
  // 6. ctx_new (streams encoder_out, 104.8 MB; deterministic two-stage)
  k_ctx_partial<<<dim3(BB, 8), 256, 0, stream>>>(a_out, enc_out, part_ctx);
  k_ctx_reduce<<<BB * TWOH / 4 / 256, 256, 0, stream>>>(part_ctx, ctxn);
  // 7. p_gen + pack A2 = bf16([h | ctx_new])
  k_pgen_pack<<<BB, 256, 0, stream>>>(ctxn, hid, ctx_out_f, W_pg, b_pg, pgen, a2_bf);
  // 8. out1 = A2 @ W_o1^T + b_o1                 (WMMA, N=512, K=1536; + bf16 copy)
  k_gemm_bf16<<<4, 256, 0, stream>>>(a2_bf, W_o1, b_o1, out1_f, out1_bf,
                                     HH, 3 * HH, HH, 0);
  // 9. logits = out1 @ W_o2^T + b_o2 into vd rows (WMMA + async-LDS A staging)
  k_gemm_bf16_lds<<<(VV / 16 + 7) / 8, 256, 0, stream>>>(out1_bf, W_o2, b_o2, vd,
                                                         VV, LDVD);
  // 10. vocab softmax * p_gen, zero OOV tail (float4 passes)
  k_softmax_v<<<BB, 256, 0, stream>>>(vd, pgen);
  // 11. scatter-add copy distribution
  k_scatter<<<(SS * BB + 255) / 256, 256, 0, stream>>>(a_out, texts, pgen, vd);
}
